// Multihead_attention_5377299055284
// MI455X (gfx1250) — matrix operand order repaired and checked by reading, not compiled, not run on hardware
//
#include <hip/hip_runtime.h>
#include <hip/hip_bf16.h>

#define D_MODEL 1024
#define N_HEAD  16
#define HEAD_DIM 64
#define BATCH   4
#define SEQ     2048
#define ROWS    (BATCH*SEQ)        // 8192

typedef __bf16 bf16_t;
typedef __attribute__((ext_vector_type(16))) __bf16 v16bf;
typedef __attribute__((ext_vector_type(8)))  __bf16 v8bf;
typedef __attribute__((ext_vector_type(8)))  float  v8f;

// ---------------------------------------------------------------------------
// WMMA helpers (CDNA5: V_WMMA_F32_16X16X32_BF16, wave32)
// ---------------------------------------------------------------------------
static __device__ __forceinline__ v8f wmma_bf16(v16bf a, v16bf b, v8f c) {
  return __builtin_amdgcn_wmma_f32_16x16x32_bf16(false, a, false, b, (short)0, c,
                                                 false, false);
}

// A-matrix 16x32 bf16 fragment, row contiguous in memory over K.
// ISA layout: lanes 0-15 hold K={0..7,16..23}; lanes 16-31 hold K={8..15,24..31}.
static __device__ __forceinline__ v16bf load_frag_a(const bf16_t* rowk0, int half) {
  v8bf lo = *reinterpret_cast<const v8bf*>(rowk0 + half * 8);
  v8bf hi = *reinterpret_cast<const v8bf*>(rowk0 + 16 + half * 8);
  return __builtin_shufflevector(lo, hi, 0, 1, 2, 3, 4, 5, 6, 7,
                                 8, 9, 10, 11, 12, 13, 14, 15);
}

// B-matrix 32x16 bf16 fragment where column n of B is contiguous over K.
// Layout: lanes 0-15 hold K={0..7,16..23}; lanes 16-31 hold K={8..15,24..31}.
static __device__ __forceinline__ v16bf load_frag_b(const bf16_t* colk0, int half) {
  v8bf lo = *reinterpret_cast<const v8bf*>(colk0 + half * 8);
  v8bf hi = *reinterpret_cast<const v8bf*>(colk0 + 16 + half * 8);
  return __builtin_shufflevector(lo, hi, 0, 1, 2, 3, 4, 5, 6, 7,
                                 8, 9, 10, 11, 12, 13, 14, 15);
}

// ---------------------------------------------------------------------------
// f32 -> bf16 conversion
// ---------------------------------------------------------------------------
__global__ void cvt_f32_bf16(const float* __restrict__ in,
                             bf16_t* __restrict__ out, int n) {
  int i = blockIdx.x * 256 + threadIdx.x;
  if (i < n) out[i] = (bf16_t)in[i];
}

// ---------------------------------------------------------------------------
// GEMM: out = A(MxK bf16, row-major) @ W^T (W is NxK bf16, row-major) + bias
// Block: 256 threads (8 waves). Wave computes 16 rows x 64 cols.
// Grid: (M/128, N/64).
// MODE: 0 = f32 row-major out, 1 = bf16 row-major out,
//       2 = bf16 V-transposed out: Vt[((b*16+h)*64+d)*SEQ + t]  (vectorized store)
// ---------------------------------------------------------------------------
template <int MODE>
__global__ __launch_bounds__(256) void gemm_xwT(const bf16_t* __restrict__ A,
                                                const bf16_t* __restrict__ W,
                                                const float* __restrict__ bias,
                                                void* __restrict__ out,
                                                int M, int N, int K) {
  const int wave = threadIdx.x >> 5;
  const int lane = threadIdx.x & 31;
  const int n0   = lane & 15;
  const int half = lane >> 4;
  const int rBase = blockIdx.x * 128 + wave * 16;
  const int cBase = blockIdx.y * 64;

  const bf16_t* aRow = A + (size_t)(rBase + n0) * K;
  const bf16_t* wRow[4];
#pragma unroll
  for (int j = 0; j < 4; ++j)
    wRow[j] = W + (size_t)(cBase + j * 16 + n0) * K;

  v8f acc[4] = {};
#pragma unroll 2
  for (int k0 = 0; k0 < K; k0 += 32) {
    v16bf a = load_frag_a(aRow + k0, half);
#pragma unroll
    for (int j = 0; j < 4; ++j)
      acc[j] = wmma_bf16(a, load_frag_b(wRow[j] + k0, half), acc[j]);
  }

#pragma unroll
  for (int j = 0; j < 4; ++j) {
    const int col = cBase + j * 16 + n0;
    const float bb = bias[col];
    if (MODE == 2) {
      // V-transposed: per lane the 8 accum rows are consecutive t -> one b128 store
      const int b = rBase >> 11;          // / SEQ
      const int t0 = (rBase & (SEQ - 1)) + half * 8;
      const int h = col >> 6, d = col & 63;
      v8bf vv;
#pragma unroll
      for (int i = 0; i < 8; ++i) vv[i] = (bf16_t)(acc[j][i] + bb);
      *reinterpret_cast<v8bf*>((bf16_t*)out +
          ((size_t)(b * N_HEAD + h) * HEAD_DIM + d) * SEQ + t0) = vv;
    } else {
#pragma unroll
      for (int i = 0; i < 8; ++i) {
        const int row = rBase + half * 8 + i;
        const float v = acc[j][i] + bb;
        if (MODE == 1)
          ((bf16_t*)out)[(size_t)row * N + col] = (bf16_t)v;
        else
          ((float*)out)[(size_t)row * N + col] = v;
      }
    }
  }
}

// ---------------------------------------------------------------------------
// Attention pass A: per key-column stats for the query-axis softmax.
// colM[k] = max_{q>=k} s(q,k),  colZ[k] = sum_{q>=k} exp(s(q,k)-colM[k])
// Grid: (B*H, SEQ/128); block 256 (8 waves, each owns 16 key columns).
// Diagonal tile handled with masking once; the rest streams branchlessly with
// double-buffered Q fragments so loads overlap WMMA + stats VALU.
// ---------------------------------------------------------------------------
__global__ __launch_bounds__(256) void attn_colstats(const bf16_t* __restrict__ Q,
                                                     const bf16_t* __restrict__ K,
                                                     float* __restrict__ colM,
                                                     float* __restrict__ colZ) {
  const int wave = threadIdx.x >> 5;
  const int lane = threadIdx.x & 31;
  const int n0   = lane & 15;
  const int half = lane >> 4;
  const int bh = blockIdx.x, b = bh >> 4, h = bh & 15;
  const int kBase = blockIdx.y * 128 + wave * 16;
  const int kcol  = kBase + n0;
  const float NEG_INF = -__builtin_inff();

  const bf16_t* Qb = Q + (size_t)b * SEQ * D_MODEL + h * HEAD_DIM;
  const bf16_t* Kb = K + (size_t)b * SEQ * D_MODEL + h * HEAD_DIM;
  const bf16_t* kRow = Kb + (size_t)kcol * D_MODEL;
  const v16bf kf0 = load_frag_b(kRow,      half);
  const v16bf kf1 = load_frag_b(kRow + 32, half);

  const float scale = 0.125f;  // 1/sqrt(64)
  float runM, runZ;

  // ---- diagonal tile (the only masked one) ----
  {
    const bf16_t* qRow = Qb + (size_t)(kBase + n0) * D_MODEL;
    v8f s = {};
    s = wmma_bf16(load_frag_a(qRow,      half), kf0, s);
    s = wmma_bf16(load_frag_a(qRow + 32, half), kf1, s);
    float tmax = NEG_INF;
    float pv[8];
#pragma unroll
    for (int i = 0; i < 8; ++i) {
      const int q = kBase + half * 8 + i;
      pv[i] = s[i] * scale;
      tmax = fmaxf(tmax, (q >= kcol) ? pv[i] : NEG_INF);
    }
    float tsum = 0.f;
#pragma unroll
    for (int i = 0; i < 8; ++i) {
      const int q = kBase + half * 8 + i;
      tsum += (q >= kcol) ? __expf(pv[i] - tmax) : 0.f;
    }
    runM = tmax;   // may be -inf for lanes whose 8 rows are all masked
    runZ = tsum;   // 0 in that case
  }

  // ---- stream remaining (fully unmasked) tiles, branchless update ----
  int qb = kBase + 16;
  if (qb < SEQ) {
    const bf16_t* qRow = Qb + (size_t)(qb + n0) * D_MODEL;
    v16bf a0 = load_frag_a(qRow,      half);
    v16bf a1 = load_frag_a(qRow + 32, half);
    for (; qb < SEQ; qb += 16) {
      const int qn = (qb + 16 < SEQ) ? qb + 16 : qb;  // clamp for prefetch
      const bf16_t* qRowN = Qb + (size_t)(qn + n0) * D_MODEL;
      v16bf p0 = load_frag_a(qRowN,      half);
      v16bf p1 = load_frag_a(qRowN + 32, half);

      v8f s = {};
      s = wmma_bf16(a0, kf0, s);
      s = wmma_bf16(a1, kf1, s);

      float tmax = s[0] * scale;
#pragma unroll
      for (int i = 1; i < 8; ++i) tmax = fmaxf(tmax, s[i] * scale);
      float tsum = 0.f;
#pragma unroll
      for (int i = 0; i < 8; ++i) tsum += __expf(s[i] * scale - tmax);

      const float nM = fmaxf(runM, tmax);
      runZ = runZ * __expf(runM - nM) + tsum * __expf(tmax - nM);
      runM = nM;
      a0 = p0;
      a1 = p1;
    }
  }

  // combine the two half-lanes holding the same column (rows 0-7 vs 8-15)
  const float oM = __shfl_xor(runM, 16, 32);
  const float oZ = __shfl_xor(runZ, 16, 32);
  const float Mx = fmaxf(runM, oM);   // finite: every column has >=1 live row
  const float Z  = runZ * __expf(runM - Mx) + oZ * __expf(oM - Mx);
  if (half == 0) {
    colM[(size_t)bh * SEQ + kcol] = Mx;
    colZ[(size_t)bh * SEQ + kcol] = Z;
  }
}

// ---------------------------------------------------------------------------
// Attention pass B: ctx = P @ V, with P[q,k] = exp(s-colM[k])/colZ[k] (causal).
// Grid: (B*H, SEQ/128); block 256 (8 waves, each one 16-row q tile x 64 d).
// V is consumed in transposed layout Vt[(bh*64+d)*SEQ + t] -> contiguous b128
// B-fragments. P round-trips LDS (C-layout -> A-layout). Exactly one k-chunk
// (kb = (qBase/32)*32) needs causal masking; earlier chunks are compare-free.
// ---------------------------------------------------------------------------
__global__ __launch_bounds__(256) void attn_ctx(const bf16_t* __restrict__ Q,
                                                const bf16_t* __restrict__ K,
                                                const bf16_t* __restrict__ Vt,
                                                const float* __restrict__ colM,
                                                const float* __restrict__ colZ,
                                                bf16_t* __restrict__ Ctx) {
  __shared__ bf16_t lds[8 * 16 * 32];
  const int wave = threadIdx.x >> 5;
  const int lane = threadIdx.x & 31;
  const int n0   = lane & 15;
  const int half = lane >> 4;
  const int bh = blockIdx.x, b = bh >> 4, h = bh & 15;
  const int qBase = blockIdx.y * 128 + wave * 16;

  const bf16_t* Qb  = Q  + (size_t)b * SEQ * D_MODEL + h * HEAD_DIM;
  const bf16_t* Kb  = K  + (size_t)b * SEQ * D_MODEL + h * HEAD_DIM;
  const bf16_t* Vtb = Vt + (size_t)bh * HEAD_DIM * SEQ;
  const float* cm = colM + (size_t)bh * SEQ;
  const float* cz = colZ + (size_t)bh * SEQ;
  bf16_t* myLds = lds + wave * (16 * 32);

  const bf16_t* qRow = Qb + (size_t)(qBase + n0) * D_MODEL;
  const v16bf qf0 = load_frag_a(qRow,      half);
  const v16bf qf1 = load_frag_a(qRow + 32, half);

  const float scale = 0.125f;
  v8f acc[4] = {};

  auto do_chunk = [&](int kb, bool masked) {
    // S tiles: 16 q-rows x 32 k-cols (two 16x16 accumulators)
    const bf16_t* kRow0 = Kb + (size_t)(kb + n0) * D_MODEL;
    const bf16_t* kRow1 = kRow0 + (size_t)16 * D_MODEL;
    v8f s0 = {}, s1 = {};
    s0 = wmma_bf16(qf0, load_frag_b(kRow0,      half), s0);
    s0 = wmma_bf16(qf1, load_frag_b(kRow0 + 32, half), s0);
    s1 = wmma_bf16(qf0, load_frag_b(kRow1,      half), s1);
    s1 = wmma_bf16(qf1, load_frag_b(kRow1 + 32, half), s1);

    // p = exp(s - m[k]) / Z[k]; write bf16 P tile to LDS
#pragma unroll
    for (int j = 0; j < 2; ++j) {
      const v8f s = j ? s1 : s0;
      const int kidx = kb + j * 16 + n0;
      const float mj = cm[kidx];
      const float rz = 1.f / cz[kidx];
#pragma unroll
      for (int i = 0; i < 8; ++i) {
        float p = __expf(s[i] * scale - mj) * rz;
        if (masked) {
          const int q = qBase + half * 8 + i;
          p = (kidx <= q) ? p : 0.f;
        }
        myLds[(half * 8 + i) * 32 + j * 16 + n0] = (bf16_t)p;
      }
    }
    __asm__ volatile("s_wait_dscnt 0" ::: "memory");

    // Re-read P as an A-layout fragment (row n0, K over 32)
    const bf16_t* lp = myLds + n0 * 32;
    v8bf plo = *reinterpret_cast<const v8bf*>(lp + half * 8);
    v8bf phi = *reinterpret_cast<const v8bf*>(lp + 16 + half * 8);
    v16bf pf = __builtin_shufflevector(plo, phi, 0, 1, 2, 3, 4, 5, 6, 7,
                                       8, 9, 10, 11, 12, 13, 14, 15);

    // ctx += P (16x32) @ V (32x64); Vt columns are contiguous over k
#pragma unroll
    for (int j2 = 0; j2 < 4; ++j2) {
      v16bf vf = load_frag_b(Vtb + (size_t)(j2 * 16 + n0) * SEQ + kb, half);
      acc[j2] = wmma_bf16(pf, vf, acc[j2]);
    }
    __asm__ volatile("s_wait_dscnt 0" ::: "memory");  // LDS reuse next chunk
  };

  const int kbFull = (qBase >> 5) << 5;           // chunks [0, kbFull) unmasked
  for (int kb = 0; kb < kbFull; kb += 32) do_chunk(kb, false);
  do_chunk(kbFull, true);                          // exactly one masked chunk

  bf16_t* outRow = Ctx + (size_t)(b * SEQ + qBase) * D_MODEL + h * HEAD_DIM;
#pragma unroll
  for (int j2 = 0; j2 < 4; ++j2)
#pragma unroll
    for (int i = 0; i < 8; ++i)
      outRow[(size_t)(half * 8 + i) * D_MODEL + j2 * 16 + n0] =
          (bf16_t)acc[j2][i];
}

// ---------------------------------------------------------------------------
// Host launcher
// ---------------------------------------------------------------------------
extern "C" void kernel_launch(void* const* d_in, const int* in_sizes, int n_in,
                              void* d_out, int out_size, void* d_ws, size_t ws_size,
                              hipStream_t stream) {
  const float* q  = (const float*)d_in[0];
  const float* Wq = (const float*)d_in[3];
  const float* bq = (const float*)d_in[4];
  const float* Wk = (const float*)d_in[5];
  const float* bk = (const float*)d_in[6];
  const float* Wv = (const float*)d_in[7];
  const float* bv = (const float*)d_in[8];
  const float* Wo = (const float*)d_in[9];
  const float* bo = (const float*)d_in[10];
  float* out = (float*)d_out;

  char* ws = (char*)d_ws;
  size_t off = 0;
  auto alloc = [&](size_t bytes) -> void* {
    void* p = ws + off;
    off += (bytes + 255) & ~(size_t)255;
    return p;
  };

  const size_t ACT = (size_t)ROWS * D_MODEL;      // 8,388,608 elements
  const size_t WEL = (size_t)D_MODEL * D_MODEL;   // 1,048,576 elements

  bf16_t* Xb  = (bf16_t*)alloc(ACT * 2);
  bf16_t* Wqb = (bf16_t*)alloc(WEL * 2);
  bf16_t* Wkb = (bf16_t*)alloc(WEL * 2);
  bf16_t* Wvb = (bf16_t*)alloc(WEL * 2);
  bf16_t* Wob = (bf16_t*)alloc(WEL * 2);
  bf16_t* Qb  = (bf16_t*)alloc(ACT * 2);
  bf16_t* Kb  = (bf16_t*)alloc(ACT * 2);
  bf16_t* Vtb = (bf16_t*)alloc(ACT * 2);          // transposed V
  bf16_t* Cb  = (bf16_t*)alloc(ACT * 2);
  float*  cM  = (float*)alloc((size_t)BATCH * N_HEAD * SEQ * 4);
  float*  cZ  = (float*)alloc((size_t)BATCH * N_HEAD * SEQ * 4);

  // 1) bf16 conversions
  cvt_f32_bf16<<<(int)((ACT + 255) / 256), 256, 0, stream>>>(q, Xb, (int)ACT);
  cvt_f32_bf16<<<(int)((WEL + 255) / 256), 256, 0, stream>>>(Wq, Wqb, (int)WEL);
  cvt_f32_bf16<<<(int)((WEL + 255) / 256), 256, 0, stream>>>(Wk, Wkb, (int)WEL);
  cvt_f32_bf16<<<(int)((WEL + 255) / 256), 256, 0, stream>>>(Wv, Wvb, (int)WEL);
  cvt_f32_bf16<<<(int)((WEL + 255) / 256), 256, 0, stream>>>(Wo, Wob, (int)WEL);

  // 2) Q/K/V projections (reference applies all three to q);
  //    V is written directly in per-(b,h) transposed layout.
  dim3 ggrid(ROWS / 128, D_MODEL / 64);
  gemm_xwT<1><<<ggrid, 256, 0, stream>>>(Xb, Wqb, bq, Qb,  ROWS, D_MODEL, D_MODEL);
  gemm_xwT<1><<<ggrid, 256, 0, stream>>>(Xb, Wkb, bk, Kb,  ROWS, D_MODEL, D_MODEL);
  gemm_xwT<2><<<ggrid, 256, 0, stream>>>(Xb, Wvb, bv, Vtb, ROWS, D_MODEL, D_MODEL);

  // 3) attention (query-axis softmax, causal)
  dim3 agrid(BATCH * N_HEAD, SEQ / 128);
  attn_colstats<<<agrid, 256, 0, stream>>>(Qb, Kb, cM, cZ);
  attn_ctx<<<agrid, 256, 0, stream>>>(Qb, Kb, Vtb, cM, cZ, Cb);

  // 4) output projection (f32 out)
  gemm_xwT<0><<<ggrid, 256, 0, stream>>>(Cb, Wob, bo, out, ROWS, D_MODEL, D_MODEL);
}
